// QuantLinear_34385508172324
// MI455X (gfx1250) — matrix operand order repaired and checked by reading, not compiled, not run on hardware
//
#include <hip/hip_runtime.h>

// ---------------------------------------------------------------------------
// W4A32 quantized linear via CDNA5 WMMA (gfx1250, wave32).
//   out[M,N] = x[M,K] * dequant_int4(qweight)[K,N] + bias[N]
// GEMM: M=8192, N=11008, K=4096 (739 GFLOP) -> matrix-engine bound.
// v_wmma_f32_16x16x32_f16, f32 accumulate. LDS double buffer (dynamic LDS,
// 72 KB of the WGP's 320 KB), one barrier per K-chunk, packed-FMA dequant.
// ---------------------------------------------------------------------------

typedef _Float16 h2   __attribute__((ext_vector_type(2)));
typedef _Float16 v4h  __attribute__((ext_vector_type(4)));
typedef _Float16 v8h  __attribute__((ext_vector_type(8)));
typedef _Float16 v16h __attribute__((ext_vector_type(16)));
typedef float    v4f  __attribute__((ext_vector_type(4)));
typedef float    v8f  __attribute__((ext_vector_type(8)));

#define BM 128   // block tile M
#define BN 128   // block tile N
#define BK 64    // block tile K (two 16x16x32 WMMA k-steps)
#define LDA 72   // LDS pitch in halves (144B = 36 banks; gcd(36,64)=4 -> rows 0..15 conflict-free)
#define LDB 72

#define ABUF (BM * LDA)   // halves per A buffer
#define BBUF (BN * LDB)   // halves per B buffer
#define SMEM_BYTES ((2 * ABUF + 2 * BBUF) * 2)   // 73728 B, dynamic LDS

__global__ __launch_bounds__(256, 1)
void qlinear_wmma_f16(const float* __restrict__ x,
                      const int*   __restrict__ qw,
                      const float* __restrict__ scale,
                      const float* __restrict__ bias,
                      float*       __restrict__ out,
                      int M, int N, int K) {
  extern __shared__ __align__(16) _Float16 smem[];
  _Float16* As = smem;                 // two A buffers back-to-back
  _Float16* Bs = smem + 2 * ABUF;      // two B buffers back-to-back

  const int t    = threadIdx.x;
  const int lane = t & 31;
  const int hl   = lane & 15;          // half-lane index
  const bool lo  = (lane < 16);
  const int wave = t >> 5;             // 8 waves
  const int wm   = wave >> 2;          // 0..1 : 64-row slab
  const int wn   = wave & 3;           // 0..3 : 32-col slab

  const int m0 = blockIdx.y * BM;
  const int n0 = blockIdx.x * BN;

  // ---- staging registers: global -> regs (prefetch) -> LDS ----
  v4f      areg[8];    // 8 x float4 of x-tile per thread
  unsigned qreg[16];   // 16 packed int4-pairs per thread
  float    sreg;       // one scale per thread per K-chunk

  const int arow = t >> 4;      // A: base row (step +16)
  const int ac4  = t & 15;      // A: float4 column within BK
  const int bn   = t & 127;     // B: weight column owned by this thread
  const int bkr  = t >> 7;      // B: 0/1 -> which 16 qweight rows (=32 k values)

  const int NKC = K / BK;

  auto load_global = [&](int kc) {
    const int k0 = kc * BK;
    const float* xrow = x + (size_t)m0 * K + k0;
#pragma unroll
    for (int i = 0; i < 8; ++i)
      areg[i] = *(const v4f*)(xrow + (size_t)(arow + 16 * i) * K + ac4 * 4);
    const int* qp = qw + (size_t)(k0 / 2 + bkr * 16) * N + (n0 + bn);
#pragma unroll
    for (int i = 0; i < 16; ++i)
      qreg[i] = (unsigned)qp[(size_t)i * N];
    sreg = scale[(size_t)(k0 / 32 + bkr) * N + (n0 + bn)];
  };

  auto store_lds = [&](int buf) {
    _Float16* a = As + buf * ABUF;
    _Float16* b = Bs + buf * BBUF;
    // x tile: f32 -> f16, row-major [m][k]
#pragma unroll
    for (int i = 0; i < 8; ++i) {
      v4h h;
      h.x = (_Float16)areg[i].x;
      h.y = (_Float16)areg[i].y;
      h.z = (_Float16)areg[i].z;
      h.w = (_Float16)areg[i].w;
      *(v4h*)(&a[(arow + 16 * i) * LDA + ac4 * 4]) = h;
    }
    // weight tile: exponent-bias dequant.  0x6400|v is f16 (1024+v) exactly,
    // so w = (1024+v)*s + (-1032*s) = (v-8)*s  -> one packed FMA per 2 nibbles.
    h2 s2, c2;
    s2.x = s2.y = (_Float16)sreg;
    c2.x = c2.y = (_Float16)(-1032.0f * sreg);
    h2 wb[16];
#pragma unroll
    for (int i = 0; i < 16; ++i) {
      const unsigned q    = qreg[i];
      const unsigned bits = 0x64006400u | (q & 0xFu) | ((q & 0xF0u) << 12);
      wb[i] = __builtin_bit_cast(h2, bits) * s2 + c2;   // v_pk_fma_f16
    }
#pragma unroll
    for (int j = 0; j < 4; ++j)
      *(v8h*)(&b[bn * LDB + bkr * 32 + 8 * j]) = ((const v8h*)wb)[j];
  };

  v8f acc[4][2] = {};  // wave computes 4x2 tiles of 16x16

  auto compute = [&](int buf) {
    const _Float16* a = As + buf * ABUF;
    const _Float16* b = Bs + buf * BBUF;
#pragma unroll
    for (int ks = 0; ks < BK; ks += 32) {
      // A frags (ISA 16-bit A 16x32): lane<16 -> K {0..7,16..23}, lane>=16 ->
      // K {8..15,24..31}, row M=hl: two 16B ds loads each.
      const int aoff = lo ? 0 : 8;
      v16h af[4];
#pragma unroll
      for (int mt = 0; mt < 4; ++mt) {
        const _Float16* p = &a[(wm * 64 + mt * 16 + hl) * LDA + ks + aoff];
        v16h av;
        ((v8h*)&av)[0] = *(const v8h*)(p);
        ((v8h*)&av)[1] = *(const v8h*)(p + 16);
        af[mt] = av;
      }
      // B frags (32x16): lane<16 -> K {0..7,16..23}, lane>=16 -> K {8..15,24..31}, col N=hl.
      const int boff = lo ? 0 : 8;
      v16h bf[2];
#pragma unroll
      for (int nt = 0; nt < 2; ++nt) {
        const _Float16* p = &b[(wn * 32 + nt * 16 + hl) * LDB + ks + boff];
        v16h bv;
        ((v8h*)&bv)[0] = *(const v8h*)(p);
        ((v8h*)&bv)[1] = *(const v8h*)(p + 16);
        bf[nt] = bv;
      }
#pragma unroll
      for (int mt = 0; mt < 4; ++mt)
#pragma unroll
        for (int nt = 0; nt < 2; ++nt)
          acc[mt][nt] = __builtin_amdgcn_wmma_f32_16x16x32_f16(
              false, af[mt], false, bf[nt], (short)0, acc[mt][nt], false, false);
    }
  };

  // ---- double-buffered pipeline: one barrier per K-chunk ----
  load_global(0);
  store_lds(0);
  if (NKC > 1) load_global(1);
  __syncthreads();
  for (int kc = 0; kc < NKC; ++kc) {
    const int cur = kc & 1;
    if (kc + 1 < NKC) store_lds(cur ^ 1);   // fill other buffer (reads done last iter)
    if (kc + 2 < NKC) load_global(kc + 2);  // VMEM overlaps WMMA below
    compute(cur);
    __syncthreads();                        // reads of cur + writes of nxt complete
  }

  // ---- epilogue: C/D layout (v8f: lane<16 rows r, lane>=16 rows 8+r) ----
#pragma unroll
  for (int mt = 0; mt < 4; ++mt) {
#pragma unroll
    for (int nt = 0; nt < 2; ++nt) {
      const int col  = n0 + wn * 32 + nt * 16 + hl;
      const float bv = bias[col];
      const int rb   = m0 + wm * 64 + mt * 16 + (lo ? 0 : 8);
#pragma unroll
      for (int r = 0; r < 8; ++r)
        out[(size_t)(rb + r) * N + col] = acc[mt][nt][r] + bv;
    }
  }
}

extern "C" void kernel_launch(void* const* d_in, const int* in_sizes, int n_in,
                              void* d_out, int out_size, void* d_ws, size_t ws_size,
                              hipStream_t stream) {
  const float* x     = (const float*)d_in[0];
  const int*   qw    = (const int*)d_in[1];
  const float* scale = (const float*)d_in[2];
  const float* bias  = (const float*)d_in[3];
  float*       out   = (float*)d_out;

  const int N = in_sizes[3];                 // 11008
  const int K = 2 * (in_sizes[1] / N);       // 4096
  const int M = in_sizes[0] / K;             // 8192

  dim3 grid(N / BN, M / BM);                 // 86 x 64
  qlinear_wmma_f16<<<grid, 256, SMEM_BYTES, stream>>>(x, qw, scale, bias, out,
                                                      M, N, K);
}
